// Down_42468636623217
// MI455X (gfx1250) — matrix operand order repaired and checked by reading, not compiled, not run on hardware
//
#include <hip/hip_runtime.h>
#include <hip/hip_bf16.h>
#include <cstdint>

// ---------------------------------------------------------------------------
// Types for WMMA fragments (wave32, v_wmma_f32_16x16x32_f16)
// ---------------------------------------------------------------------------
typedef __attribute__((ext_vector_type(16))) _Float16 v16h;
typedef __attribute__((ext_vector_type(8)))  _Float16 v8h;
typedef __attribute__((ext_vector_type(8)))  float    v8f;

typedef unsigned int uint32x4 __attribute__((ext_vector_type(4)));
typedef int          int32x8  __attribute__((ext_vector_type(8)));
typedef int          int32x4  __attribute__((ext_vector_type(4)));

#if defined(__has_builtin)
#  if __has_builtin(__builtin_amdgcn_tensor_load_to_lds)
#    define USE_TDM 1
#  endif
#endif
#ifndef USE_TDM
#  define USE_TDM 0
#endif

__device__ __forceinline__ v8f wmma16x16x32(v16h a, v16h b, v8f c) {
    return __builtin_amdgcn_wmma_f32_16x16x32_f16(
        /*neg_a=*/false, a, /*neg_b=*/false, b,
        /*c_mod=*/(short)0, c, /*reuse_a=*/false, /*reuse_b=*/false);
}

__device__ __forceinline__ float gelu_exact(float v) {
    return 0.5f * v * (1.0f + erff(v * 0.7071067811865475f));
}

// A fragment: 16(M) x 32(K) f16, source row-major [*, lda].
__device__ __forceinline__ v16h load_a_frag(const _Float16* A, int lda, int m0, int k0, int lane) {
    int row = lane & 15;
    int kg  = (lane >> 4) * 8;
    const _Float16* p = A + (size_t)(m0 + row) * lda + k0 + kg;
    v8h lo = *(const v8h*)p;
    v8h hi = *(const v8h*)(p + 16);
    v16h r;
#pragma unroll
    for (int i = 0; i < 8; ++i) { r[i] = lo[i]; r[8 + i] = hi[i]; }
    return r;
}

// B fragment: source is weight-style row-major [N, ldb] (=B^T): contiguous per lane.
__device__ __forceinline__ v16h load_b_frag(const _Float16* B, int ldb, int n0, int k0, int lane) {
    int n  = lane & 15;
    int kg = (lane >> 4) * 8;
    const _Float16* p = B + (size_t)(n0 + n) * ldb + k0 + kg;
    v8h lo = *(const v8h*)p;
    v8h hi = *(const v8h*)(p + 16);
    v16h r;
#pragma unroll
    for (int i = 0; i < 8; ++i) { r[i] = lo[i]; r[8 + i] = hi[i]; }
    return r;
}

// Same but reading from LDS with a row pitch (in halves).
__device__ __forceinline__ v16h load_b_frag_lds(const _Float16* Bl, int pitch, int n0, int k0, int lane) {
    int n  = lane & 15;
    int kg = (lane >> 4) * 8;
    const _Float16* p = Bl + (size_t)(n0 + n) * pitch + k0 + kg;
    v8h lo = *(const v8h*)p;
    v8h hi = *(const v8h*)(p + 16);
    v16h r;
#pragma unroll
    for (int i = 0; i < 8; ++i) { r[i] = lo[i]; r[8 + i] = hi[i]; }
    return r;
}

// ---------------------------------------------------------------------------
// Shared epilogue
// ---------------------------------------------------------------------------
enum { EPI_F32 = 0, EPI_BIAS_BOTH = 1, EPI_GELU_F16 = 2, EPI_GELU_SUB_F16 = 3,
       EPI_GELU_F32 = 4, EPI_ADD_F32 = 5, EPI_SIG_F32 = 6, EPI_RELU_F32 = 7 };

template <int EPI>
__device__ __forceinline__ void epilogue_store(
    const v8f (&acc)[2][2], int bm, int bn, int lane,
    const float* __restrict__ bias, float scale,
    float* __restrict__ Cf, _Float16* __restrict__ Ch, size_t coff, int ldc,
    const _Float16* __restrict__ extraH, const float* __restrict__ extraF)
{
    int n  = lane & 15;
    int mh = (lane >> 4) * 8;
#pragma unroll
    for (int mt = 0; mt < 2; ++mt) {
#pragma unroll
        for (int nt = 0; nt < 2; ++nt) {
            int col  = bn + nt * 16 + n;
            float bv = bias ? bias[col] : 0.0f;
#pragma unroll
            for (int r = 0; r < 8; ++r) {
                int rowi   = bm + mt * 16 + mh + r;
                size_t idx = coff + (size_t)rowi * ldc + col;
                float v    = acc[mt][nt][r] * scale + bv;
                if (EPI == EPI_F32)            { Cf[idx] = v; }
                else if (EPI == EPI_BIAS_BOTH) { Cf[idx] = v; Ch[idx] = (_Float16)v; }
                else if (EPI == EPI_GELU_F16)  { Ch[idx] = (_Float16)gelu_exact(v); }
                else if (EPI == EPI_GELU_SUB_F16) {
                    Ch[idx] = (_Float16)(gelu_exact(v) - (float)extraH[idx]);
                }
                else if (EPI == EPI_GELU_F32)  { Cf[idx] = gelu_exact(v); }
                else if (EPI == EPI_ADD_F32)   { Cf[idx] = extraF[idx] + v; }
                else if (EPI == EPI_SIG_F32)   { Cf[idx] = 1.0f / (1.0f + expf(-v)); }
                else if (EPI == EPI_RELU_F32)  { Cf[idx] = v > 0.0f ? v : 0.0f; }
            }
        }
    }
}

// ---------------------------------------------------------------------------
// Generic batched NT GEMM: 8 waves, block tile 128M x 64N, wave tile 32x32.
// ---------------------------------------------------------------------------
template <int EPI>
__global__ __launch_bounds__(256) void gemm_nt_kernel(
    const _Float16* __restrict__ A, int lda, long astride,
    const _Float16* __restrict__ B, int ldb, long bstride,
    const float* __restrict__ bias, float scale,
    float* __restrict__ Cf, _Float16* __restrict__ Ch, long cstride, int ldc,
    const _Float16* __restrict__ extraH, const float* __restrict__ extraF, int K)
{
    int z = blockIdx.z;
    A += (size_t)z * astride;
    B += (size_t)z * bstride;
    size_t coff = (size_t)z * cstride;

    int lane = threadIdx.x & 31;
    int wid  = threadIdx.x >> 5;
    int mw   = wid & 3;
    int nw   = wid >> 2;
    int bm   = blockIdx.x * 128 + mw * 32;
    int bn   = blockIdx.y * 64  + nw * 32;

    v8f acc[2][2] = {};
    for (int k0 = 0; k0 < K; k0 += 32) {
        v16h a0 = load_a_frag(A, lda, bm,      k0, lane);
        v16h a1 = load_a_frag(A, lda, bm + 16, k0, lane);
        v16h b0 = load_b_frag(B, ldb, bn,      k0, lane);
        v16h b1 = load_b_frag(B, ldb, bn + 16, k0, lane);
        acc[0][0] = wmma16x16x32(a0, b0, acc[0][0]);
        acc[0][1] = wmma16x16x32(a0, b1, acc[0][1]);
        acc[1][0] = wmma16x16x32(a1, b0, acc[1][0]);
        acc[1][1] = wmma16x16x32(a1, b1, acc[1][1]);
    }
    epilogue_store<EPI>(acc, bm, bn, lane, bias, scale, Cf, Ch, coff, ldc, extraH, extraF);
}

// ---------------------------------------------------------------------------
// FFN GEMM (M=65536, N=K=128): B (128x128 f16 weights) staged into LDS via the
// Tensor Data Mover (async DMA, TENSORcnt) with 272B row pitch for conflict-free
// ds_load_b128 fragment reads. Fallback: cooperative copy.
// ---------------------------------------------------------------------------
template <int EPI>
__global__ __launch_bounds__(256) void gemm_nt_ldsb_kernel(
    const _Float16* __restrict__ A,
    const _Float16* __restrict__ B,
    const float* __restrict__ bias, float scale,
    float* __restrict__ Cf, _Float16* __restrict__ Ch,
    const _Float16* __restrict__ extraH, const float* __restrict__ extraF)
{
    __shared__ _Float16 ldsW[128 * 136];    // pitch 136 halves = 272 B

#if USE_TDM
    if (threadIdx.x == 0) {
        uint64_t ga = (uint64_t)(uintptr_t)B;
        uint32_t la = (uint32_t)(uintptr_t)(void*)&ldsW[0];
        uint32x4 g0;
        g0[0] = 1u;                                   // count=1, user mode
        g0[1] = la;                                   // lds_addr
        g0[2] = (uint32_t)ga;                         // global_addr[31:0]
        g0[3] = (uint32_t)((ga >> 32) & 0x01FFFFFFull) | 0x80000000u; // addr[56:32] | type=2
        int32x8 g1;
        g1[0] = (int)((1u << 16)      // data_size = 2 bytes
                    | (1u << 20)      // pad_enable
                    | (5u << 22)      // pad_interval = 64 dwords (256B)
                    | (3u << 25));    // pad_amount = 4 dwords (16B)
        g1[1] = (int)(128u << 16);    // tensor_dim0 = 128 (low16)
        g1[2] = (int)(128u << 16);    // tensor_dim0 hi=0 | tensor_dim1 = 128 (low16)
        g1[3] = (int)(128u << 16);    // tensor_dim1 hi=0 | tile_dim0 = 128
        g1[4] = 128;                  // tile_dim1 = 128, tile_dim2 = 0
        g1[5] = 128;                  // tensor_dim0_stride low32 = 128
        g1[6] = (int)(16384u << 16);  // stride0 hi=0 | tensor_dim1_stride low16
        g1[7] = 0;                    // tensor_dim1_stride hi
        int32x4 g2 = {0, 0, 0, 0};
        int32x4 g3 = {0, 0, 0, 0};
        int32x8 g4 = {0, 0, 0, 0, 0, 0, 0, 0};
        __builtin_amdgcn_tensor_load_to_lds(g0, g1, g2, g3, g4, 0);
        __builtin_amdgcn_s_wait_tensorcnt(0);
    }
    __syncthreads();
#else
    for (int i = threadIdx.x; i < 128 * 64; i += 256) {   // dword copy
        int row = i >> 6;
        int col = i & 63;
        ((uint32_t*)(ldsW + row * 136))[col] = ((const uint32_t*)(B + row * 128))[col];
    }
    __syncthreads();
#endif

    int lane = threadIdx.x & 31;
    int wid  = threadIdx.x >> 5;
    int mw   = wid & 3;
    int nw   = wid >> 2;
    int bm   = blockIdx.x * 128 + mw * 32;
    int bn   = blockIdx.y * 64  + nw * 32;

    v8f acc[2][2] = {};
#pragma unroll
    for (int k0 = 0; k0 < 128; k0 += 32) {
        v16h a0 = load_a_frag(A, 128, bm,      k0, lane);
        v16h a1 = load_a_frag(A, 128, bm + 16, k0, lane);
        v16h b0 = load_b_frag_lds(ldsW, 136, bn,      k0, lane);
        v16h b1 = load_b_frag_lds(ldsW, 136, bn + 16, k0, lane);
        acc[0][0] = wmma16x16x32(a0, b0, acc[0][0]);
        acc[0][1] = wmma16x16x32(a0, b1, acc[0][1]);
        acc[1][0] = wmma16x16x32(a1, b0, acc[1][0]);
        acc[1][1] = wmma16x16x32(a1, b1, acc[1][1]);
    }
    epilogue_store<EPI>(acc, bm, bn, lane, bias, scale, Cf, Ch, 0, 128, extraH, extraF);
}

// ---------------------------------------------------------------------------
// Implicit-GEMM 4x4/stride2/pad1 conv: dw = relu(conv(x, l1_w) + l1_b)
// M=65536 tokens, N=128, K=2048 (ci*16 + kh*4 + kw). Branch-free im2col:
// per-lane clamped offsets + validity bitmask computed once, loads unconditional.
// ---------------------------------------------------------------------------
__global__ __launch_bounds__(256) void conv_dw_kernel(
    const float* __restrict__ x, const _Float16* __restrict__ W,
    const float* __restrict__ bias, float* __restrict__ out)
{
    int lane = threadIdx.x & 31;
    int wid  = threadIdx.x >> 5;
    int mw   = wid & 3;
    int nw   = wid >> 2;
    int bm   = blockIdx.x * 128 + mw * 32;
    int bn   = blockIdx.y * 64  + nw * 32;
    int row  = lane & 15;
    int kg   = (lane >> 4) * 8;    // 0 or 8

    int      offs[2][16];
    unsigned okbits[2];
    int      xbase[2];
#pragma unroll
    for (int t = 0; t < 2; ++t) {
        int tok = bm + t * 16 + row;
        int b = tok >> 12, rem = tok & 4095, h2 = rem >> 6, w2 = rem & 63;
        xbase[t] = b << 21;                       // b * 128 * 16384
        unsigned ok = 0;
#pragma unroll
        for (int kh = 0; kh < 4; ++kh) {
#pragma unroll
            for (int kw = 0; kw < 4; ++kw) {
                int iy  = 2 * h2 - 1 + kh;
                int ix  = 2 * w2 - 1 + kw;
                int iyc = iy < 0 ? 0 : (iy > 127 ? 127 : iy);
                int ixc = ix < 0 ? 0 : (ix > 127 ? 127 : ix);
                offs[t][kh * 4 + kw] = (iyc << 7) + ixc;
                if ((unsigned)iy < 128u && (unsigned)ix < 128u) ok |= 1u << (kh * 4 + kw);
            }
        }
        okbits[t] = ok;
    }

    v8f acc[2][2] = {};
    for (int k0 = 0; k0 < 2048; k0 += 32) {
        // prefetch planes 2 iterations ahead
        if (k0 + 64 < 2048)
            __builtin_prefetch(&x[(size_t)xbase[0] + ((size_t)((k0 >> 4) + 4) << 14) + offs[0][0]], 0, 1);

        v16h a[2];
#pragma unroll
        for (int t = 0; t < 2; ++t) {
#pragma unroll
            for (int g = 0; g < 2; ++g) {
                int ci = (k0 >> 4) + g;
                const float* xp = x + (size_t)xbase[t] + ((size_t)ci << 14);
#pragma unroll
                for (int u = 0; u < 8; ++u) {
                    int rk   = kg + u;                    // 0..15
                    float v  = xp[offs[t][rk]];           // unconditional load
                    bool  ok = (okbits[t] >> rk) & 1u;
                    a[t][g * 8 + u] = (_Float16)(ok ? v : 0.0f);   // v_cndmask
                }
            }
        }
        v16h b0 = load_b_frag(W, 2048, bn,      k0, lane);
        v16h b1 = load_b_frag(W, 2048, bn + 16, k0, lane);
        acc[0][0] = wmma16x16x32(a[0], b0, acc[0][0]);
        acc[0][1] = wmma16x16x32(a[0], b1, acc[0][1]);
        acc[1][0] = wmma16x16x32(a[1], b0, acc[1][0]);
        acc[1][1] = wmma16x16x32(a[1], b1, acc[1][1]);
    }

    int n = lane & 15, mh = (lane >> 4) * 8;
#pragma unroll
    for (int mt = 0; mt < 2; ++mt)
#pragma unroll
        for (int nt = 0; nt < 2; ++nt) {
            int col = bn + nt * 16 + n;
            float bv = bias[col];
#pragma unroll
            for (int r = 0; r < 8; ++r) {
                int rowi = bm + mt * 16 + mh + r;
                float v = acc[mt][nt][r] + bv;
                out[(size_t)rowi * 128 + col] = v > 0.0f ? v : 0.0f;
            }
        }
}

// ---------------------------------------------------------------------------
// Haar DWT -> Q=LL [T,128], KL=[HL|LH] [T,256], Vt=HH transposed [128][128][512].
// ---------------------------------------------------------------------------
__global__ __launch_bounds__(128) void dwt_kernel(
    const float* __restrict__ x, _Float16* __restrict__ Q,
    _Float16* __restrict__ Vt, _Float16* __restrict__ KL)
{
    int tok = blockIdx.x;
    int c   = threadIdx.x;
    int b   = tok >> 12;
    int rem = tok & 4095;
    int h2  = rem >> 6;
    int w2  = rem & 63;
    const float* p = x + (((size_t)(b * 128 + c)) << 14) + (h2 << 8) + (w2 << 1);
    float a  = p[0];
    float bb = p[1];
    float cc = p[128];
    float dd = p[129];
    float ll   = (a + bb + cc + dd) * 0.5f;
    float lohi = (a - bb + cc - dd) * 0.5f;   // y_HL
    float hilo = (a + bb - cc - dd) * 0.5f;   // y_LH
    float hihi = (a - bb - cc + dd) * 0.5f;   // y_HH
    Q[(size_t)tok * 128 + c] = (_Float16)ll;
    int z = tok >> 9, kk = tok & 511;
    Vt[((size_t)z * 128 + c) * 512 + kk] = (_Float16)hihi;
    KL[(size_t)tok * 256 + c]       = (_Float16)lohi;
    KL[(size_t)tok * 256 + 128 + c] = (_Float16)hilo;
}

// ---------------------------------------------------------------------------
// Column softmax stats over q for S stored (chunk, q, k): online max/sum.
// ---------------------------------------------------------------------------
__global__ __launch_bounds__(256) void att_stats_kernel(
    const float* __restrict__ S, float2* __restrict__ stats)
{
    int g = blockIdx.x * blockDim.x + threadIdx.x;   // 128 chunks * 512 k
    int z = g >> 9, k = g & 511;
    const float* col = S + (size_t)z * 512 * 512 + k;
    float m = -1e30f, d = 0.0f;
    for (int q = 0; q < 512; ++q) {
        float v = col[(size_t)q * 512];
        if (v > m) { d = d * expf(m - v) + 1.0f; m = v; }
        else       { d += expf(v - m); }
    }
    float2 r; r.x = m; r.y = d;
    stats[g] = r;
}

// ---------------------------------------------------------------------------
// P@V: O[q,c] = sum_k exp(S[q,k]-m[k])/d[k] * V[k,c]; B frags from transposed Vt.
// ---------------------------------------------------------------------------
__device__ __forceinline__ v16h load_a_pv(const float* __restrict__ S,
                                          const float2* __restrict__ st,
                                          int m0, int k0, int lane)
{
    int row = lane & 15;
    int kg  = (lane >> 4) * 8;
    const float* p = S + (size_t)(m0 + row) * 512 + k0;
    v16h r;
#pragma unroll
    for (int g = 0; g < 2; ++g) {
        int kb = g * 16 + kg;
#pragma unroll
        for (int u = 0; u < 8; ++u) {
            float2 md = st[k0 + kb + u];
            r[g * 8 + u] = (_Float16)(expf(p[kb + u] - md.x) / md.y);
        }
    }
    return r;
}

__global__ __launch_bounds__(256) void att_pv_kernel(
    const float* __restrict__ S, const float2* __restrict__ stats,
    const _Float16* __restrict__ Vt, float* __restrict__ O)
{
    int z = blockIdx.z;
    const float*    Sz = S + (size_t)z * 512 * 512;
    const float2*   st = stats + (size_t)z * 512;
    const _Float16* Vz = Vt + (size_t)z * 128 * 512;    // [c][k]
    float* Oz = O + (size_t)z * 512 * 128;

    int lane = threadIdx.x & 31;
    int wid  = threadIdx.x >> 5;
    int mw   = wid & 3;
    int nw   = wid >> 2;
    int bm   = blockIdx.x * 128 + mw * 32;
    int bn   = blockIdx.y * 64  + nw * 32;

    v8f acc[2][2] = {};
    for (int k0 = 0; k0 < 512; k0 += 32) {
        v16h a0 = load_a_pv(Sz, st, bm,      k0, lane);
        v16h a1 = load_a_pv(Sz, st, bm + 16, k0, lane);
        v16h b0 = load_b_frag(Vz, 512, bn,      k0, lane);  // Vt rows are c
        v16h b1 = load_b_frag(Vz, 512, bn + 16, k0, lane);
        acc[0][0] = wmma16x16x32(a0, b0, acc[0][0]);
        acc[0][1] = wmma16x16x32(a0, b1, acc[0][1]);
        acc[1][0] = wmma16x16x32(a1, b0, acc[1][0]);
        acc[1][1] = wmma16x16x32(a1, b1, acc[1][1]);
    }
    int n = lane & 15, mh = (lane >> 4) * 8;
#pragma unroll
    for (int mt = 0; mt < 2; ++mt)
#pragma unroll
        for (int nt = 0; nt < 2; ++nt)
#pragma unroll
            for (int r = 0; r < 8; ++r) {
                int rowi = bm + mt * 16 + mh + r;
                Oz[(size_t)rowi * 128 + bn + nt * 16 + n] = acc[mt][nt][r];
            }
}

// ---------------------------------------------------------------------------
// FFN pre-pass: per-token LN, emit A1=ln1(x), A3=ln3(x), Ax=x (f16)
// ---------------------------------------------------------------------------
__global__ __launch_bounds__(128) void ffn_pre_kernel(
    const float* __restrict__ X,
    const float* __restrict__ g1, const float* __restrict__ b1,
    const float* __restrict__ g3, const float* __restrict__ b3,
    _Float16* __restrict__ A1, _Float16* __restrict__ A3, _Float16* __restrict__ Ax)
{
    __shared__ float red[128];
    int tok = blockIdx.x, c = threadIdx.x;
    size_t idx = (size_t)tok * 128 + c;
    float v = X[idx];

    red[c] = v; __syncthreads();
    for (int s = 64; s > 0; s >>= 1) { if (c < s) red[c] += red[c + s]; __syncthreads(); }
    float m = red[0] * (1.0f / 128.0f); __syncthreads();

    float dv = v - m;
    red[c] = dv * dv; __syncthreads();
    for (int s = 64; s > 0; s >>= 1) { if (c < s) red[c] += red[c + s]; __syncthreads(); }
    float var = red[0] * (1.0f / 128.0f);

    float xn = dv * rsqrtf(var + 1e-5f);
    A1[idx] = (_Float16)(xn * g1[c] + b1[c]);
    A3[idx] = (_Float16)(xn * g3[c] + b3[c]);
    Ax[idx] = (_Float16)v;
}

// ---------------------------------------------------------------------------
// Final: out = gate * ln(x2+x3) * (bilinear(x,64,64) - dw) + y_img, NCHW f32.
// ---------------------------------------------------------------------------
__global__ __launch_bounds__(256) void final_kernel(
    const float* __restrict__ X23, const float* __restrict__ GATE,
    const float* __restrict__ dw, const float* __restrict__ Y,
    const float* __restrict__ x,
    const float* __restrict__ gn, const float* __restrict__ bn,
    float* __restrict__ out)
{
    __shared__ float rs1[8][32];
    __shared__ float rs2[8][32];
    __shared__ float mB[32];
    __shared__ float vB[32];

    int w   = threadIdx.x & 31;
    int tc  = threadIdx.x >> 5;
    int bid = blockIdx.x;               // 16*64*2
    int b   = bid >> 7;
    int rem = bid & 127;
    int h2  = rem >> 1;
    int wh  = rem & 1;
    int w2  = wh * 32 + w;
    int tok = b * 4096 + h2 * 64 + w2;
    size_t tb = (size_t)tok * 128;

    float s1 = 0.0f, s2 = 0.0f;
    for (int ci = tc; ci < 128; ci += 8) {
        float v = X23[tb + ci];
        s1 += v; s2 += v * v;
    }
    rs1[tc][w] = s1; rs2[tc][w] = s2;
    __syncthreads();
    if (tc == 0) {
        float a = 0.0f, q = 0.0f;
        for (int i = 0; i < 8; ++i) { a += rs1[i][w]; q += rs2[i][w]; }
        float m = a * (1.0f / 128.0f);
        float var = q * (1.0f / 128.0f) - m * m;
        mB[w] = m;
        vB[w] = rsqrtf(var + 1e-5f);
    }
    __syncthreads();
    float m = mB[w], rv = vB[w];

    float py = (float)h2 * (127.0f / 63.0f);
    int   y0 = (int)floorf(py);
    int   y1 = y0 + 1 < 128 ? y0 + 1 : 127;
    float wy = py - (float)y0;
    float px = (float)w2 * (127.0f / 63.0f);
    int   x0 = (int)floorf(px);
    int   x1 = x0 + 1 < 128 ? x0 + 1 : 127;
    float wx = px - (float)x0;

    for (int ci = tc; ci < 128; ci += 8) {
        float v  = X23[tb + ci];
        float a  = GATE[tb + ci] * ((v - m) * rv * gn[ci] + bn[ci]);
        const float* xc = x + (((size_t)(b * 128 + ci)) << 14);
        float r0 = xc[(y0 << 7) + x0] * (1.0f - wx) + xc[(y0 << 7) + x1] * wx;
        float r1 = xc[(y1 << 7) + x0] * (1.0f - wx) + xc[(y1 << 7) + x1] * wx;
        float d2 = r0 * (1.0f - wy) + r1 * wy;
        float o  = a * (d2 - dw[tb + ci]) + Y[tb + ci];
        out[(((size_t)(b * 128 + ci)) << 12) + (h2 << 6) + w2] = o;
    }
}

// ---------------------------------------------------------------------------
// f32 -> f16 weight conversion
// ---------------------------------------------------------------------------
__global__ void cvt_f16_kernel(const float* __restrict__ s, _Float16* __restrict__ d, int n) {
    int i = blockIdx.x * blockDim.x + threadIdx.x;
    if (i < n) d[i] = (_Float16)s[i];
}

// ---------------------------------------------------------------------------
// Host orchestration
// ---------------------------------------------------------------------------
extern "C" void kernel_launch(void* const* d_in, const int* in_sizes, int n_in,
                              void* d_out, int out_size, void* d_ws, size_t ws_size,
                              hipStream_t stream) {
    const float* x       = (const float*)d_in[0];
    const float* conv1_w = (const float*)d_in[1];
    const float* conv1_b = (const float*)d_in[2];
    const float* l1_w    = (const float*)d_in[3];
    const float* l1_b    = (const float*)d_in[4];
    const float* ln1_g   = (const float*)d_in[5];
    const float* ln1_b   = (const float*)d_in[6];
    const float* fc1_w   = (const float*)d_in[7];
    const float* fc1_b   = (const float*)d_in[8];
    const float* div_w   = (const float*)d_in[9];
    const float* div_b   = (const float*)d_in[10];
    const float* fc2_w   = (const float*)d_in[11];
    const float* fc2_b   = (const float*)d_in[12];
    const float* ln3_g   = (const float*)d_in[13];
    const float* ln3_b   = (const float*)d_in[14];
    const float* fc3a_w  = (const float*)d_in[15];
    const float* fc3a_b  = (const float*)d_in[16];
    const float* fc3b_w  = (const float*)d_in[17];
    const float* fc3b_b  = (const float*)d_in[18];
    const float* lnn_g   = (const float*)d_in[19];
    const float* lnn_b   = (const float*)d_in[20];
    const float* act_w   = (const float*)d_in[21];
    const float* act_b   = (const float*)d_in[22];
    float* out = (float*)d_out;

    char* base = (char*)d_ws;
    const size_t MB = 1ull << 20;
    _Float16* Qh   = (_Float16*)(base + 0 * MB);     // 16 MB [T,128] LL
    _Float16* Vt   = (_Float16*)(base + 16 * MB);    // 16 MB [128][128][512] HH^T
    _Float16* K16  = (_Float16*)(base + 32 * MB);    // 16 MB [T,128]
    float*    Ybuf = (float*)   (base + 48 * MB);    // 32 MB [T,128]
    _Float16* KL   = (_Float16*)(base + 80 * MB);    // 32 MB [T,256]; reused:
    _Float16* T1   = (_Float16*)(base + 80 * MB);
    _Float16* T2   = (_Float16*)(base + 96 * MB);
    float*    S    = (float*)   (base + 112 * MB);   // 128 MB; reused after PV:
    _Float16* A1   = (_Float16*)(base + 112 * MB);
    _Float16* A3   = (_Float16*)(base + 128 * MB);
    _Float16* Axh  = (_Float16*)(base + 144 * MB);
    float*    X23  = (float*)   (base + 160 * MB);
    float*    GATE = (float*)   (base + 192 * MB);
    float*    ATTX = (float*)   (base + 240 * MB);
    float*    DW   = (float*)   (base + 272 * MB);
    float2*   STAT = (float2*)  (base + 304 * MB);
    char*     wts  = base + 305 * MB;
    _Float16* w1_16   = (_Float16*)(wts + 0);
    _Float16* fc1_16  = (_Float16*)(wts + 64 * 1024);
    _Float16* div_16  = (_Float16*)(wts + 96 * 1024);
    _Float16* fc2_16  = (_Float16*)(wts + 128 * 1024);
    _Float16* fc3a_16 = (_Float16*)(wts + 160 * 1024);
    _Float16* fc3b_16 = (_Float16*)(wts + 192 * 1024);
    _Float16* act_16  = (_Float16*)(wts + 224 * 1024);
    _Float16* l1_16   = (_Float16*)(wts + 256 * 1024);

    cvt_f16_kernel<<<128, 256, 0, stream>>>(conv1_w, w1_16, 32768);
    cvt_f16_kernel<<<64, 256, 0, stream>>>(fc1_w,  fc1_16,  16384);
    cvt_f16_kernel<<<64, 256, 0, stream>>>(div_w,  div_16,  16384);
    cvt_f16_kernel<<<64, 256, 0, stream>>>(fc2_w,  fc2_16,  16384);
    cvt_f16_kernel<<<64, 256, 0, stream>>>(fc3a_w, fc3a_16, 16384);
    cvt_f16_kernel<<<64, 256, 0, stream>>>(fc3b_w, fc3b_16, 16384);
    cvt_f16_kernel<<<64, 256, 0, stream>>>(act_w,  act_16,  16384);
    cvt_f16_kernel<<<1024, 256, 0, stream>>>(l1_w, l1_16, 262144);

    dwt_kernel<<<65536, 128, 0, stream>>>(x, Qh, Vt, KL);

    gemm_nt_kernel<EPI_BIAS_BOTH><<<dim3(512, 2, 1), 256, 0, stream>>>(
        KL, 256, 0, w1_16, 256, 0, conv1_b, 1.0f, Ybuf, K16, 0, 128,
        nullptr, nullptr, 256);

    conv_dw_kernel<<<dim3(512, 2, 1), 256, 0, stream>>>(x, l1_16, l1_b, DW);

    gemm_nt_kernel<EPI_F32><<<dim3(4, 8, 128), 256, 0, stream>>>(
        Qh, 128, 512 * 128, K16, 128, 512 * 128, nullptr, 0.3535533906f,
        S, nullptr, 512L * 512, 512, nullptr, nullptr, 128);

    att_stats_kernel<<<256, 256, 0, stream>>>(S, STAT);

    att_pv_kernel<<<dim3(4, 2, 128), 256, 0, stream>>>(S, STAT, Vt, ATTX);

    ffn_pre_kernel<<<65536, 128, 0, stream>>>(ATTX, ln1_g, ln1_b, ln3_g, ln3_b, A1, A3, Axh);

    gemm_nt_ldsb_kernel<EPI_GELU_F16><<<dim3(512, 2, 1), 256, 0, stream>>>(
        A1, fc1_16, fc1_b, 1.0f, nullptr, T1, nullptr, nullptr);
    gemm_nt_ldsb_kernel<EPI_GELU_SUB_F16><<<dim3(512, 2, 1), 256, 0, stream>>>(
        Axh, div_16, div_b, 1.0f, nullptr, T2, T1, nullptr);
    gemm_nt_ldsb_kernel<EPI_GELU_F32><<<dim3(512, 2, 1), 256, 0, stream>>>(
        T2, fc2_16, fc2_b, 1.0f, X23, nullptr, nullptr, nullptr);
    gemm_nt_ldsb_kernel<EPI_GELU_F16><<<dim3(512, 2, 1), 256, 0, stream>>>(
        A3, fc3a_16, fc3a_b, 1.0f, nullptr, T1, nullptr, nullptr);
    gemm_nt_ldsb_kernel<EPI_ADD_F32><<<dim3(512, 2, 1), 256, 0, stream>>>(
        T1, fc3b_16, fc3b_b, 1.0f, X23, nullptr, nullptr, X23);
    gemm_nt_ldsb_kernel<EPI_SIG_F32><<<dim3(512, 2, 1), 256, 0, stream>>>(
        Axh, act_16, act_b, 1.0f, GATE, nullptr, nullptr, nullptr);

    final_kernel<<<2048, 256, 0, stream>>>(X23, GATE, DW, Ybuf, x, lnn_g, lnn_b, out);

    (void)in_sizes; (void)n_in; (void)out_size; (void)ws_size;
}
